// AdvancedTransformerBlock_15006615733156
// MI455X (gfx1250) — matrix operand order repaired and checked by reading, not compiled, not run on hardware
//
#include <hip/hip_runtime.h>
#include <hip/hip_bf16.h>
#include <math.h>

// ---------------------------------------------------------------------------
// Types for CDNA5 WMMA (wave32): v_wmma_f32_16x16x32_bf16
// ---------------------------------------------------------------------------
typedef __attribute__((ext_vector_type(16))) __bf16 v16bf;
typedef __attribute__((ext_vector_type(8)))  __bf16 v8bf;
typedef __attribute__((ext_vector_type(2)))  __bf16 v2bf;
typedef __attribute__((ext_vector_type(8)))  float  v8f;

#define B_SZ 2
#define N_SZ 2048
#define D_SZ 1024
#define H_SZ 16
#define HD_SZ 64
#define WINDOW 256

__device__ __forceinline__ v8f wmma_bf16(v16bf a, v16bf b, v8f c) {
    // (neg_a, A, neg_b, B, c_mod, C, reuse_a, reuse_b)
    return __builtin_amdgcn_wmma_f32_16x16x32_bf16(false, a, false, b, (short)0, c, false, false);
}

__device__ __forceinline__ void splitbf(float x, __bf16& h, __bf16& l) {
    h = (__bf16)x;
    l = (__bf16)(x - (float)h);
}

// A-matrix fragment (16x32 bf16): lane<16 -> K {kb..kb+7, kb+16..kb+23}, kb=0
//                                 lane>=16 -> kb=8
__device__ __forceinline__ v16bf loadA_frag(const __bf16* rowp, int kb) {
    v8bf a = *(const v8bf*)(rowp + kb);
    v8bf b = *(const v8bf*)(rowp + kb + 16);
    v16bf r;
#pragma unroll
    for (int i = 0; i < 8; ++i) { r[i] = a[i]; r[i + 8] = b[i]; }
    return r;
}

// B-matrix fragment (32x16 bf16): lane<16 -> K {0..7, 16..23}; lane>=16 -> K {8..15, 24..31}
__device__ __forceinline__ v16bf loadB_frag(const __bf16* colp, int kbB) {
    v8bf a = *(const v8bf*)(colp + kbB);
    v8bf b = *(const v8bf*)(colp + kbB + 16);
    v16bf r;
#pragma unroll
    for (int i = 0; i < 8; ++i) { r[i] = a[i]; r[i + 8] = b[i]; }
    return r;
}

// ---------------------------------------------------------------------------
// LayerNorm: one 256-thread block per row, D = 1024 (4 floats / thread)
// ---------------------------------------------------------------------------
__global__ __launch_bounds__(256) void ln_kernel(const float* __restrict__ x,
                                                 const float* __restrict__ g,
                                                 const float* __restrict__ bt,
                                                 float* __restrict__ y) {
    __shared__ float s1[256], s2[256];
    const int row = blockIdx.x;
    const int tid = threadIdx.x;
    const float* xr = x + (size_t)row * D_SZ;
    float4 xv = *(const float4*)(xr + tid * 4);
    float a  = xv.x + xv.y + xv.z + xv.w;
    float sq = xv.x * xv.x + xv.y * xv.y + xv.z * xv.z + xv.w * xv.w;
    s1[tid] = a; s2[tid] = sq;
    __syncthreads();
#pragma unroll
    for (int s = 128; s > 0; s >>= 1) {
        if (tid < s) { s1[tid] += s1[tid + s]; s2[tid] += s2[tid + s]; }
        __syncthreads();
    }
    const float mu  = s1[0] * (1.0f / D_SZ);
    const float var = s2[0] * (1.0f / D_SZ) - mu * mu;
    const float rstd = rsqrtf(var + 1e-5f);
    float4 gv = *(const float4*)(g + tid * 4);
    float4 bv = *(const float4*)(bt + tid * 4);
    float4 ov;
    ov.x = (xv.x - mu) * rstd * gv.x + bv.x;
    ov.y = (xv.y - mu) * rstd * gv.y + bv.y;
    ov.z = (xv.z - mu) * rstd * gv.z + bv.z;
    ov.w = (xv.w - mu) * rstd * gv.w + bv.w;
    *(float4*)(y + (size_t)row * D_SZ + tid * 4) = ov;
}

// ---------------------------------------------------------------------------
// Split-bf16 GEMM: C[M,N] = act(A[M,K] @ W[K,N] + bias) (+ resid)
// Block tile 128x128, 8 waves (4x2), wave tile 32x64, K chunk 32.
// A·W ~= Ah·Wh + Ah·Wl + Al·Wh  (3 WMMAs per 16x16x32 step, near-fp32 accuracy)
// ACT: 0 = none, 1 = exact GELU.  HAS_RES: add resid[] before store.
// ---------------------------------------------------------------------------
#define LSTR 40  // padded K-stride (bf16 elems): 80B rows -> bank-conflict-free, 16B aligned

template <int ACT, bool HAS_RES>
__global__ __launch_bounds__(256) void gemm_bf16x2(const float* __restrict__ A,
                                                   const float* __restrict__ W,
                                                   const float* __restrict__ bias,
                                                   const float* __restrict__ resid,
                                                   float* __restrict__ C,
                                                   int M, int N, int K) {
    __shared__ __bf16 Ahi[128 * LSTR];
    __shared__ __bf16 Alo[128 * LSTR];
    __shared__ __bf16 Whi[128 * LSTR];
    __shared__ __bf16 Wlo[128 * LSTR];

    const int tid  = threadIdx.x;
    const int lane = tid & 31;
    const int wave = tid >> 5;
    const int wm = wave >> 1;      // 0..3
    const int wn = wave & 1;       // 0..1
    const int m0 = blockIdx.y * 128;
    const int n0 = blockIdx.x * 128;
    const int lrow = lane & 15;
    const int kbA = (lane < 16) ? 0 : 8;
    const int kbB = (lane < 16) ? 0 : 8;

    v8f acc[2][4];
#pragma unroll
    for (int mi = 0; mi < 2; ++mi)
#pragma unroll
        for (int ni = 0; ni < 4; ++ni)
#pragma unroll
            for (int r = 0; r < 8; ++r) acc[mi][ni][r] = 0.0f;

    for (int k0 = 0; k0 < K; k0 += 32) {
        __syncthreads();
        // ---- stage A tile: 128 rows x 32 k (f32 -> bf16 hi/lo), packed b32 stores ----
        {
            const int r  = tid >> 3;         // 0..31
            const int kq = (tid & 7) * 4;    // 0..28
#pragma unroll
            for (int p = 0; p < 4; ++p) {
                const int row = r + p * 32;
                float4 v = *(const float4*)(A + (size_t)(m0 + row) * K + k0 + kq);
                float vv[4] = {v.x, v.y, v.z, v.w};
                __bf16 h[4], l[4];
#pragma unroll
                for (int j = 0; j < 4; ++j) splitbf(vv[j], h[j], l[j]);
                __bf16* ph = &Ahi[row * LSTR + kq];
                __bf16* pl = &Alo[row * LSTR + kq];
                *(v2bf*)(ph)     = (v2bf){h[0], h[1]};
                *(v2bf*)(ph + 2) = (v2bf){h[2], h[3]};
                *(v2bf*)(pl)     = (v2bf){l[0], l[1]};
                *(v2bf*)(pl + 2) = (v2bf){l[2], l[3]};
            }
        }
        // ---- stage W tile: 32 k x 128 n, transposed to [n][k] ----
        {
            const int kk = tid >> 5;         // 0..7
            const int nc = (tid & 31) * 4;   // 0..124
#pragma unroll
            for (int p = 0; p < 4; ++p) {
                const int kr = kk + p * 8;
                float4 v = *(const float4*)(W + (size_t)(k0 + kr) * N + n0 + nc);
                float vv[4] = {v.x, v.y, v.z, v.w};
#pragma unroll
                for (int j = 0; j < 4; ++j) {
                    __bf16 h, l; splitbf(vv[j], h, l);
                    Whi[(nc + j) * LSTR + kr] = h;
                    Wlo[(nc + j) * LSTR + kr] = l;
                }
            }
        }
        __syncthreads();

        v16bf ah[2], al[2], bh[4], bl[4];
#pragma unroll
        for (int mi = 0; mi < 2; ++mi) {
            const int row = wm * 32 + mi * 16 + lrow;
            ah[mi] = loadA_frag(Ahi + row * LSTR, kbA);
            al[mi] = loadA_frag(Alo + row * LSTR, kbA);
        }
#pragma unroll
        for (int ni = 0; ni < 4; ++ni) {
            const int col = wn * 64 + ni * 16 + lrow;
            bh[ni] = loadB_frag(Whi + col * LSTR, kbB);
            bl[ni] = loadB_frag(Wlo + col * LSTR, kbB);
        }
#pragma unroll
        for (int mi = 0; mi < 2; ++mi)
#pragma unroll
            for (int ni = 0; ni < 4; ++ni) {
                acc[mi][ni] = wmma_bf16(ah[mi], bh[ni], acc[mi][ni]);
                acc[mi][ni] = wmma_bf16(ah[mi], bl[ni], acc[mi][ni]);
                acc[mi][ni] = wmma_bf16(al[mi], bh[ni], acc[mi][ni]);
            }
    }

    // ---- epilogue: bias + activation + residual (fully specialized) ----
    const int rbase = (lane < 16) ? 0 : 8;
#pragma unroll
    for (int mi = 0; mi < 2; ++mi) {
#pragma unroll
        for (int ni = 0; ni < 4; ++ni) {
            const int col = n0 + wn * 64 + ni * 16 + lrow;
            const float bv = bias[col];
#pragma unroll
            for (int r = 0; r < 8; ++r) {
                const int mrow = m0 + wm * 32 + mi * 16 + rbase + r;
                float v = acc[mi][ni][r] + bv;
                if (ACT == 1) v = 0.5f * v * (1.0f + erff(v * 0.70710678118654752f));
                if (HAS_RES) v += resid[(size_t)mrow * N + col];
                C[(size_t)mrow * N + col] = v;
            }
        }
    }
}

// ---------------------------------------------------------------------------
// ALiBi sliding-window causal attention, one wave per (b, h, 16-query tile).
// bf16 WMMA for S = Q K^T and O += P V, online (flash) softmax in f32.
// qkv layout: [B, N, 3, H, HD] f32;  o layout: [B, N, H, HD] f32.
// ---------------------------------------------------------------------------
__global__ __launch_bounds__(32) void attn_kernel(const float* __restrict__ qkv,
                                                  float* __restrict__ o) {
    __shared__ float  S_lds[16 * 33];
    __shared__ __bf16 P_lds[16 * LSTR];
    __shared__ __bf16 V_lds[64 * LSTR];
    __shared__ float  alpha_lds[16];
    __shared__ float  l_lds[16];

    const int t = blockIdx.x;      // query tile
    const int h = blockIdx.y;      // head
    const int b = blockIdx.z;      // batch
    const int q0 = t * 16;
    const int lane = threadIdx.x;
    const int lrow = lane & 15;
    const int kbA = (lane < 16) ? 0 : 8;
    const int kbB = (lane < 16) ? 0 : 8;
    const int rbase = (lane < 16) ? 0 : 8;

    const float scale = 0.125f;                      // HD^-0.5
    const float slope = exp2f(-0.5f * (float)(h + 1));

    // ---- load Q fragments (hi-only bf16), 2 chunks over HD=64 ----
    v16bf qf[2];
    {
        const int qrow = q0 + lrow;
        const float* qb = qkv + ((((size_t)b * N_SZ + qrow) * 3 + 0) * H_SZ + h) * HD_SZ;
#pragma unroll
        for (int c = 0; c < 2; ++c) {
            const float* p0 = qb + c * 32 + kbA;
            float4 a0 = ((const float4*)p0)[0];
            float4 a1 = ((const float4*)p0)[1];
            float4 b0 = ((const float4*)(p0 + 16))[0];
            float4 b1 = ((const float4*)(p0 + 16))[1];
            float lo8[8] = {a0.x, a0.y, a0.z, a0.w, a1.x, a1.y, a1.z, a1.w};
            float hi8[8] = {b0.x, b0.y, b0.z, b0.w, b1.x, b1.y, b1.z, b1.w};
#pragma unroll
            for (int i = 0; i < 8; ++i) {
                qf[c][i]     = (__bf16)lo8[i];
                qf[c][i + 8] = (__bf16)hi8[i];
            }
        }
    }

    v8f O[4];
#pragma unroll
    for (int nt = 0; nt < 4; ++nt)
#pragma unroll
        for (int r = 0; r < 8; ++r) O[nt][r] = 0.0f;

    float m_run = -1e30f, l_run = 0.0f;

    int js = q0 - WINDOW;
    if (js < 0) js = 0;
    js &= ~31;

    for (int j0 = js; j0 < q0 + 16; j0 += 32) {
        // ---- K fragments: 2 key tiles x 2 d-chunks (hi-only bf16) ----
        v16bf kf[2][2];
#pragma unroll
        for (int kt = 0; kt < 2; ++kt) {
            int j = j0 + kt * 16 + lrow;
            if (j > N_SZ - 1) j = N_SZ - 1;
            const float* kb = qkv + ((((size_t)b * N_SZ + j) * 3 + 1) * H_SZ + h) * HD_SZ;
#pragma unroll
            for (int c = 0; c < 2; ++c) {
                const float* p = kb + c * 32 + kbB;
#pragma unroll
                for (int q = 0; q < 4; ++q) {
                    float4 v = ((const float4*)p)[q < 2 ? q : q + 2];
                    kf[kt][c][q * 4 + 0] = (__bf16)v.x;
                    kf[kt][c][q * 4 + 1] = (__bf16)v.y;
                    kf[kt][c][q * 4 + 2] = (__bf16)v.z;
                    kf[kt][c][q * 4 + 3] = (__bf16)v.w;
                }
            }
        }
        // ---- S = Q K^T  (two 16x16 tiles) ----
        v8f s0, s1;
#pragma unroll
        for (int r = 0; r < 8; ++r) { s0[r] = 0.0f; s1[r] = 0.0f; }
        s0 = wmma_bf16(qf[0], kf[0][0], s0);
        s0 = wmma_bf16(qf[1], kf[0][1], s0);
        s1 = wmma_bf16(qf[0], kf[1][0], s1);
        s1 = wmma_bf16(qf[1], kf[1][1], s1);

        // stage S to LDS (C layout -> row-major 16x32)
#pragma unroll
        for (int r = 0; r < 8; ++r) {
            S_lds[(rbase + r) * 33 + lrow]      = s0[r];
            S_lds[(rbase + r) * 33 + 16 + lrow] = s1[r];
        }
        __syncthreads();

        // ---- per-row online softmax (lanes 0..15 own rows) ----
        if (lane < 16) {
            const int i = q0 + lane;
            float xv[32];
            float mx = m_run;
#pragma unroll
            for (int jj = 0; jj < 32; ++jj) {
                const int j = j0 + jj;
                const int rel = j - i;
                float x;
                if (rel > 0 || rel <= -WINDOW) x = -1e30f;
                else x = S_lds[lane * 33 + jj] * scale + slope * (float)rel + 1.0f;
                xv[jj] = x;
                mx = fmaxf(mx, x);
            }
            const float alpha = __expf(m_run - mx);   // mx >= m_run, safe
            float ls = 0.0f;
#pragma unroll
            for (int jj = 0; jj < 32; jj += 2) {
                const float p0 = (xv[jj]     > -1e29f) ? __expf(xv[jj]     - mx) : 0.0f;
                const float p1 = (xv[jj + 1] > -1e29f) ? __expf(xv[jj + 1] - mx) : 0.0f;
                ls += p0 + p1;
                *(v2bf*)&P_lds[lane * LSTR + jj] = (v2bf){(__bf16)p0, (__bf16)p1};
            }
            l_run = l_run * alpha + ls;
            m_run = mx;
            alpha_lds[lane] = alpha;
        }
        // ---- stage V chunk (32 keys x 64 d), transposed [d][j] ----
        {
            int j = j0 + lane;
            if (j > N_SZ - 1) j = N_SZ - 1;
            if (j < 0) j = 0;
            const float* vb = qkv + ((((size_t)b * N_SZ + j) * 3 + 2) * H_SZ + h) * HD_SZ;
#pragma unroll
            for (int d4 = 0; d4 < 16; ++d4) {
                float4 v = ((const float4*)vb)[d4];
                V_lds[(d4 * 4 + 0) * LSTR + lane] = (__bf16)v.x;
                V_lds[(d4 * 4 + 1) * LSTR + lane] = (__bf16)v.y;
                V_lds[(d4 * 4 + 2) * LSTR + lane] = (__bf16)v.z;
                V_lds[(d4 * 4 + 3) * LSTR + lane] = (__bf16)v.w;
            }
        }
        __syncthreads();

        // ---- rescale O by per-row alpha, then O += P V ----
        float asc[8];
#pragma unroll
        for (int r = 0; r < 8; ++r) asc[r] = alpha_lds[rbase + r];
#pragma unroll
        for (int nt = 0; nt < 4; ++nt)
#pragma unroll
            for (int r = 0; r < 8; ++r) O[nt][r] *= asc[r];

        v16bf pf = loadA_frag(P_lds + lrow * LSTR, kbA);
#pragma unroll
        for (int nt = 0; nt < 4; ++nt) {
            v16bf vf = loadB_frag(V_lds + (nt * 16 + lrow) * LSTR, kbB);
            O[nt] = wmma_bf16(pf, vf, O[nt]);
        }
        __syncthreads();
    }

    // ---- epilogue: divide by l, store [B,N,H,HD] ----
    if (lane < 16) l_lds[lane] = l_run;
    __syncthreads();
    float linv[8];
#pragma unroll
    for (int r = 0; r < 8; ++r) linv[r] = 1.0f / l_lds[rbase + r];
#pragma unroll
    for (int nt = 0; nt < 4; ++nt)
#pragma unroll
        for (int r = 0; r < 8; ++r) {
            const int q = q0 + rbase + r;
            o[(((size_t)b * N_SZ + q) * H_SZ + h) * HD_SZ + nt * 16 + lrow] = O[nt][r] * linv[r];
        }
}

// ---------------------------------------------------------------------------
// Launch: LN1 -> QKV GEMM -> attention -> proj(+x) -> LN2 -> FC1(GELU) -> FC2(+x2)
// ---------------------------------------------------------------------------
extern "C" void kernel_launch(void* const* d_in, const int* in_sizes, int n_in,
                              void* d_out, int out_size, void* d_ws, size_t ws_size,
                              hipStream_t stream) {
    const float* x      = (const float*)d_in[0];
    const float* qkv_w  = (const float*)d_in[1];
    const float* qkv_b  = (const float*)d_in[2];
    const float* proj_w = (const float*)d_in[3];
    const float* proj_b = (const float*)d_in[4];
    const float* ln1_g  = (const float*)d_in[5];
    const float* ln1_b  = (const float*)d_in[6];
    const float* ln2_g  = (const float*)d_in[7];
    const float* ln2_b  = (const float*)d_in[8];
    const float* fc1_w  = (const float*)d_in[9];
    const float* fc1_b  = (const float*)d_in[10];
    const float* fc2_w  = (const float*)d_in[11];
    const float* fc2_b  = (const float*)d_in[12];
    float* out = (float*)d_out;

    char* ws = (char*)d_ws;
    const size_t M = (size_t)B_SZ * N_SZ;                 // 4096
    float* h    = (float*)ws;                             // [M, D]        16.8 MB
    float* qkvb = (float*)(ws + M * D_SZ * 4);            // [M, 3D]       50.3 MB
    float* ob   = (float*)(ws + M * D_SZ * 4 + M * 3 * D_SZ * 4);  // [M, D] 16.8 MB
    float* ff1  = qkvb;                                   // [M, 4D] aliases qkv+o (67.1 MB)

    // 1) h = LN1(x)
    ln_kernel<<<dim3(M), 256, 0, stream>>>(x, ln1_g, ln1_b, h);
    // 2) qkv = h @ qkv_w + qkv_b
    gemm_bf16x2<0, false><<<dim3(3 * D_SZ / 128, M / 128), 256, 0, stream>>>(
        h, qkv_w, qkv_b, nullptr, qkvb, (int)M, 3 * D_SZ, D_SZ);
    // 3) o = attention(qkv)
    attn_kernel<<<dim3(N_SZ / 16, H_SZ, B_SZ), 32, 0, stream>>>(qkvb, ob);
    // 4) x2 = x + o @ proj_w + proj_b   (x2 -> d_out)
    gemm_bf16x2<0, true><<<dim3(D_SZ / 128, M / 128), 256, 0, stream>>>(
        ob, proj_w, proj_b, x, out, (int)M, D_SZ, D_SZ);
    // 5) h = LN2(x2)
    ln_kernel<<<dim3(M), 256, 0, stream>>>(out, ln2_g, ln2_b, h);
    // 6) ff1 = GELU(h @ fc1_w + fc1_b)
    gemm_bf16x2<1, false><<<dim3(4 * D_SZ / 128, M / 128), 256, 0, stream>>>(
        h, fc1_w, fc1_b, nullptr, ff1, (int)M, 4 * D_SZ, D_SZ);
    // 7) out = x2 + ff1 @ fc2_w + fc2_b  (in-place residual on d_out)
    gemm_bf16x2<0, true><<<dim3(D_SZ / 128, M / 128), 256, 0, stream>>>(
        ff1, fc2_w, fc2_b, out, out, (int)M, D_SZ, 4 * D_SZ);
}
